// WorldModel_60954175864997
// MI455X (gfx1250) — matrix operand order repaired and checked by reading, not compiled, not run on hardware
//
#include <hip/hip_runtime.h>
#include <hip/hip_bf16.h>
#include <math.h>

// ---------------------------------------------------------------------------
// World-model forward for MI455X (gfx1250, wave32, WMMA).
// Dense matmuls: V_WMMA_F32_16X16X32_F16, fully templated on shape so all
// B-fragment loads are one base register + immediate offsets (no spills),
// 2 N-tiles per wave (N/16 is always even here -> no partial tiles, no
// exec-mask churn; EXEC all-1s around WMMA as the ISA requires).
// ---------------------------------------------------------------------------

typedef __attribute__((ext_vector_type(16))) _Float16 v16h;
typedef __attribute__((ext_vector_type(8)))  float    v8f;

#define B_SZ   2
#define S_SZ   64
#define ROWS   128          // B*S
#define HID    1024
#define DI     2048
#define DTR    64
#define NST    16
#define LATD   576
#define CCAT   24

__device__ __forceinline__ float dev_sigmoid(float x) { return 1.0f / (1.0f + __expf(-x)); }

// --------------------------- WMMA GEMM (templated) --------------------------
// C[128,N] = act(A[128,K] @ B[K,N] + bias) (+ residual).  ldc == N.
// ACT: 0=none 1=relu 2=softplus.  Requires N%32==0 handled by NT=2 tiles.
template <int LDA, int LDB, int KDIM, int NDIM, int ACT, bool HAS_BIAS, bool HAS_RES>
__global__ __launch_bounds__(256)
void gemm_t(const float* __restrict__ A, const float* __restrict__ Bm,
            float* __restrict__ C, const float* __restrict__ bias,
            const float* __restrict__ residual) {
    constexpr int NGRP = (NDIM / 16) / 2;         // 2 N-tiles per wave; always exact
    const int lane = threadIdx.x & 31;
    const int wave = threadIdx.x >> 5;
    const int grp  = blockIdx.x * 8 + wave;
    if (grp >= (ROWS / 16) * NGRP) return;        // wave-uniform
    const int tm   = grp / NGRP;
    const int tn0  = (grp - tm * NGRP) * 2;
    const int half = lane >> 4;                   // wave32: two 16-lane halves
    const int r    = lane & 15;
    const int m    = tm * 16 + r;

    v8f acc0 = {}, acc1 = {};
    // A fragment base: k = 16*(j>=8) + 8*half + (j&7) -> two contiguous
    // 8-float runs per lane; 16B alignment provable for all configs.
    const float* a0 = A + (size_t)m * LDA + half * 8;
    // B fragment base: element j = row (kb + 16*(j>=8) + 8*half + (j&7)), col tn*16+r.
    const float* b0 = Bm + (size_t)(half * 8) * LDB + tn0 * 16 + r;
    const float* b1 = b0 + 16;

    for (int kb = 0; kb < KDIM; kb += 32) {
        v16h af;
        {
            float4 x0 = *(const float4*)(a0);
            float4 x1 = *(const float4*)(a0 + 4);
            float4 x2 = *(const float4*)(a0 + 16);
            float4 x3 = *(const float4*)(a0 + 20);
            af[0]  = (_Float16)x0.x; af[1]  = (_Float16)x0.y;
            af[2]  = (_Float16)x0.z; af[3]  = (_Float16)x0.w;
            af[4]  = (_Float16)x1.x; af[5]  = (_Float16)x1.y;
            af[6]  = (_Float16)x1.z; af[7]  = (_Float16)x1.w;
            af[8]  = (_Float16)x2.x; af[9]  = (_Float16)x2.y;
            af[10] = (_Float16)x2.z; af[11] = (_Float16)x2.w;
            af[12] = (_Float16)x3.x; af[13] = (_Float16)x3.y;
            af[14] = (_Float16)x3.z; af[15] = (_Float16)x3.w;
        }
        v16h bf0, bf1;
#pragma unroll
        for (int j = 0; j < 16; ++j) bf0[j] = (_Float16)b0[(size_t)(j + (j >= 8 ? 8 : 0)) * LDB];
        acc0 = __builtin_amdgcn_wmma_f32_16x16x32_f16(
                   false, af, false, bf0, (short)0, acc0, false, false);
#pragma unroll
        for (int j = 0; j < 16; ++j) bf1[j] = (_Float16)b1[(size_t)(j + (j >= 8 ? 8 : 0)) * LDB];
        acc1 = __builtin_amdgcn_wmma_f32_16x16x32_f16(
                   false, af, false, bf1, (short)0, acc1, false, false);
        a0 += 32;
        b0 += (size_t)32 * LDB;
        b1 += (size_t)32 * LDB;
    }

    // C/D layout: VGPR v -> M = v + 8*half, N = lane&15
    float bias0 = 0.0f, bias1 = 0.0f;
    if (HAS_BIAS) {
        bias0 = bias[tn0 * 16 + r];
        bias1 = bias[tn0 * 16 + 16 + r];
    }
#pragma unroll
    for (int v = 0; v < 8; ++v) {
        int mm = tm * 16 + half * 8 + v;
        size_t o0 = (size_t)mm * NDIM + tn0 * 16 + r;
        float v0 = acc0[v] + bias0;
        float v1 = acc1[v] + bias1;
        if (ACT == 1) { v0 = fmaxf(v0, 0.0f); v1 = fmaxf(v1, 0.0f); }
        else if (ACT == 2) {
            v0 = (v0 > 20.0f) ? v0 : log1pf(__expf(v0));
            v1 = (v1 > 20.0f) ? v1 : log1pf(__expf(v1));
        }
        if (HAS_RES) { v0 += residual[o0]; v1 += residual[o0 + 16]; }
        C[o0]      = v0;
        C[o0 + 16] = v1;
    }
}

template <int LDA, int LDB, int KDIM, int NDIM, int ACT, bool HB, bool HR>
static inline void launch_gemm_t(const float* A, const float* Bm, float* C,
                                 const float* bias, const float* res,
                                 hipStream_t st) {
    constexpr int groups = (ROWS / 16) * ((NDIM / 16) / 2);
    constexpr int blocks = (groups + 7) / 8;      // 8 waves (256 thr) per block
    gemm_t<LDA, LDB, KDIM, NDIM, ACT, HB, HR><<<blocks, 256, 0, st>>>(A, Bm, C, bias, res);
}

// N==1 head: one wave per row, shuffle-reduced dot product.
__global__ void rowdot_kernel(const float* __restrict__ A,
                              const float* __restrict__ w,
                              const float* __restrict__ b,
                              float* __restrict__ out, int rows, int K) {
    int wv   = (blockIdx.x * blockDim.x + threadIdx.x) >> 5;
    int lane = threadIdx.x & 31;
    if (wv >= rows) return;
    const float* arow = A + (size_t)wv * K;
    float s = 0.0f;
    for (int k = lane; k < K; k += 32) s += arow[k] * w[k];
#pragma unroll
    for (int off = 16; off > 0; off >>= 1) s += __shfl_down(s, off, 32);
    if (lane == 0) out[wv] = s + b[0];
}

// --------------------------- elementwise kernels ---------------------------

__global__ void pack_states_kernel(const float* __restrict__ obs,
                                   const float* __restrict__ actions,
                                   const float* __restrict__ rewards,
                                   float* __restrict__ X) {
    int idx = blockIdx.x * blockDim.x + threadIdx.x;
    if (idx >= ROWS * HID) return;
    int col = idx & (HID - 1);
    int row = idx >> 10;
    float v = 0.0f;
    if (col < LATD)       v = obs[(size_t)row * LATD + col];
    else if (col == LATD) v = actions[row];
    else if (col == LATD + 1) v = rewards[row];
    X[idx] = v;
}

// causal depthwise conv (D_CONV=4) + bias + silu over xr = xz[:, :DI]
__global__ void conv_silu_kernel(const float* __restrict__ xz,
                                 const float* __restrict__ convW,
                                 const float* __restrict__ convB,
                                 float* __restrict__ xc) {
    int idx = blockIdx.x * blockDim.x + threadIdx.x;
    if (idx >= ROWS * DI) return;
    int c = idx & (DI - 1);
    int s = (idx >> 11) & (S_SZ - 1);
    int b = idx >> 17;
    float acc = convB[c];
#pragma unroll
    for (int k = 0; k < 4; ++k) {
        int si = s + k - 3;
        if (si >= 0)
            acc += xz[(size_t)(b * S_SZ + si) * (2 * DI) + c] * convW[c * 4 + k];
    }
    xc[idx] = acc * dev_sigmoid(acc);
}

// selective scan: 4096 independent (b, channel) recurrences, 16 states each.
__global__ void scan_kernel(const float* __restrict__ A_log,
                            const float* __restrict__ Dp,
                            const float* __restrict__ dt,
                            const float* __restrict__ xc,
                            const float* __restrict__ xz,
                            const float* __restrict__ proj,  // [128,96]
                            float* __restrict__ yb) {
    int idx = blockIdx.x * blockDim.x + threadIdx.x;
    if (idx >= B_SZ * DI) return;
    int c = idx & (DI - 1);
    int b = idx >> 11;
    float An[NST], h[NST];
#pragma unroll
    for (int n = 0; n < NST; ++n) {
        An[n] = -__expf(A_log[(size_t)c * NST + n]);
        h[n] = 0.0f;
    }
    const float Dc = Dp[c];
    for (int s = 0; s < S_SZ; ++s) {
        size_t row = (size_t)b * S_SZ + s;
        float dtv = dt[row * DI + c];
        float xcv = xc[row * DI + c];
        float zv  = xz[row * (2 * DI) + DI + c];
        const float* pr = proj + row * 96;
        float y = 0.0f;
#pragma unroll
        for (int n = 0; n < NST; ++n) {
            float dA = __expf(dtv * An[n]);
            h[n] = dA * h[n] + dtv * pr[64 + n] * xcv;
            y += h[n] * pr[80 + n];
        }
        y = (y + xcv * Dc) * (zv * dev_sigmoid(zv));
        yb[row * DI + c] = y;
    }
}

__device__ __forceinline__ float hash_uniform(unsigned x) {
    x ^= x >> 17; x *= 0xed5ad4bbu;
    x ^= x >> 11; x *= 0xac4c1b51u;
    x ^= x >> 15; x *= 0x31848babu;
    x ^= x >> 14;
    return ((float)(x >> 8) + 0.5f) * (1.0f / 16777216.0f);
}

// softmax over groups of 24, mix, write dists; ST-sample -> one-hot
__global__ void softmax_sample_kernel(const float* __restrict__ logits,
                                      float* __restrict__ dists,
                                      float* __restrict__ samples) {
    int row = blockIdx.x * blockDim.x + threadIdx.x;    // B*S*CCAT rows of 24
    if (row >= ROWS * CCAT) return;
    int bs = row / CCAT;
    int g  = row - bs * CCAT;
    size_t base = (size_t)bs * LATD + g * CCAT;
    float mx = -1e30f;
    for (int j = 0; j < CCAT; ++j) mx = fmaxf(mx, logits[base + j]);
    float e[CCAT], sum = 0.0f;
    for (int j = 0; j < CCAT; ++j) { e[j] = __expf(logits[base + j] - mx); sum += e[j]; }
    float inv = 1.0f / sum;
    float best = -1e30f; int bi = 0;
    for (int j = 0; j < CCAT; ++j) {
        float p = 0.99f * e[j] * inv + (0.01f / (float)CCAT);
        dists[base + j] = p;
        float u = hash_uniform((unsigned)(row * CCAT + j) * 0x9e3779b9u + 42u);
        float gum = -__logf(-__logf(u));
        float v = __logf(p) + gum;
        if (v > best) { best = v; bi = j; }
    }
    for (int j = 0; j < CCAT; ++j) samples[base + j] = (j == bi) ? 1.0f : 0.0f;
}

// transposed conv, stride 2, kernel 4x4, SAME (pad 1)
__global__ void deconv_kernel(const float* __restrict__ in,
                              const float* __restrict__ w,   // [Co,Ci,4,4]
                              const float* __restrict__ bias,
                              float* __restrict__ out,
                              int Ci, int Co, int H, int W, int act) {
    int OH = 2 * H, OW = 2 * W;
    long total = (long)ROWS * Co * OH * OW;
    long idx = (long)blockIdx.x * blockDim.x + threadIdx.x;
    if (idx >= total) return;
    int ox = idx % OW;
    int oy = (idx / OW) % OH;
    int o  = (idx / ((long)OW * OH)) % Co;
    int n  = idx / ((long)OW * OH * Co);
    float acc = bias[o];
#pragma unroll
    for (int ky = 0; ky < 4; ++ky) {
        int t = oy + 1 - ky;
        if (t < 0 || (t & 1)) continue;
        int iy = t >> 1;
        if (iy >= H) continue;
#pragma unroll
        for (int kx = 0; kx < 4; ++kx) {
            int u = ox + 1 - kx;
            if (u < 0 || (u & 1)) continue;
            int ix = u >> 1;
            if (ix >= W) continue;
            for (int i = 0; i < Ci; ++i)
                acc += in[(((size_t)n * Ci + i) * H + iy) * W + ix]
                     * w[(((size_t)o * Ci + i) * 4 + ky) * 4 + kx];
        }
    }
    out[idx] = (act == 1) ? fmaxf(acc, 0.0f) : dev_sigmoid(acc);
}

__global__ void copy_kernel(const float* __restrict__ src, float* __restrict__ dst, int n) {
    int i = blockIdx.x * blockDim.x + threadIdx.x;
    if (i < n) dst[i] = src[i];
}

// --------------------------- host-side orchestration -----------------------

struct MambaP {
    const float *A_log, *D, *conv_W, *conv_b, *dt_W, *dt_b, *in_W, *out_W, *x_W;
};
// params pytree flattened jax.tree_leaves-style: dict keys sorted.
static inline MambaP load_mamba(void* const* d_in, int base) {
    MambaP p;
    p.A_log  = (const float*)d_in[base + 0];
    p.D      = (const float*)d_in[base + 1];
    p.conv_W = (const float*)d_in[base + 2];
    p.conv_b = (const float*)d_in[base + 3];
    p.dt_W   = (const float*)d_in[base + 4];
    p.dt_b   = (const float*)d_in[base + 5];
    p.in_W   = (const float*)d_in[base + 6];
    p.out_W  = (const float*)d_in[base + 7];
    p.x_W    = (const float*)d_in[base + 8];
    return p;
}

static void run_mamba(const MambaP& p, const float* Xin, float* Xout,
                      const float* residual,
                      float* xz, float* xc, float* proj, float* dt, float* yb,
                      hipStream_t st) {
    // xz[128,4096] = Xin @ in_W
    launch_gemm_t<HID, 2 * DI, HID, 2 * DI, 0, false, false>(Xin, p.in_W, xz, nullptr, nullptr, st);
    conv_silu_kernel<<<(ROWS * DI + 255) / 256, 256, 0, st>>>(xz, p.conv_W, p.conv_b, xc);
    // proj[128,96] = xc @ x_W
    launch_gemm_t<DI, 96, DI, 96, 0, false, false>(xc, p.x_W, proj, nullptr, nullptr, st);
    // dt[128,2048] = softplus(proj[:, :64] @ dt_W + dt_b)
    launch_gemm_t<96, DI, DTR, DI, 2, true, false>(proj, p.dt_W, dt, p.dt_b, nullptr, st);
    scan_kernel<<<(B_SZ * DI + 255) / 256, 256, 0, st>>>(p.A_log, p.D, dt, xc, xz, proj, yb);
    // Xout[128,1024] = yb @ out_W (+ residual)
    if (residual)
        launch_gemm_t<DI, HID, DI, HID, 0, false, true>(yb, p.out_W, Xout, nullptr, residual, st);
    else
        launch_gemm_t<DI, HID, DI, HID, 0, false, false>(yb, p.out_W, Xout, nullptr, nullptr, st);
}

extern "C" void kernel_launch(void* const* d_in, const int* in_sizes, int n_in,
                              void* d_out, int out_size, void* d_ws, size_t ws_size,
                              hipStream_t stream) {
    (void)in_sizes; (void)n_in; (void)out_size; (void)ws_size;
    const float* obs     = (const float*)d_in[0];
    const float* actions = (const float*)d_in[1];
    const float* rewards = (const float*)d_in[2];
    // params leaves, sorted top-level keys:
    // img_W=3, img_b=4, img_head_mamba=5..13, predictor=14..85,
    // rew_W=86, rew_b=87, reward_mamba=88..96,
    // vae: dec_W=97..99, dec_b=100..102, fc_W=103, fc_b=104
    const float* img_W = (const float*)d_in[3];
    const float* img_b = (const float*)d_in[4];
    MambaP img_p = load_mamba(d_in, 5);
    MambaP pred[8];
    for (int i = 0; i < 8; ++i) pred[i] = load_mamba(d_in, 14 + 9 * i);
    const float* rew_W = (const float*)d_in[86];
    const float* rew_b = (const float*)d_in[87];
    MambaP rew_p = load_mamba(d_in, 88);
    const float* dec_W0 = (const float*)d_in[97];
    const float* dec_W1 = (const float*)d_in[98];
    const float* dec_W2 = (const float*)d_in[99];
    const float* dec_b0 = (const float*)d_in[100];
    const float* dec_b1 = (const float*)d_in[101];
    const float* dec_b2 = (const float*)d_in[102];
    const float* fc_W   = (const float*)d_in[103];
    const float* fc_b   = (const float*)d_in[104];

    // output layout (flat, return order)
    float* out_obs   = (float*)d_out;                    // 524288
    float* out_dists = out_obs + (size_t)ROWS * 64 * 64; // 73728
    float* out_rew   = out_dists + (size_t)ROWS * LATD;  // 128
    float* out_h     = out_rew + ROWS;                   // 131072

    // workspace carve-up (floats); every offset is a multiple of 16 bytes
    float* wf = (float*)d_ws;
    float* X       = wf;             wf += (size_t)ROWS * HID;
    float* xz      = wf;             wf += (size_t)ROWS * 2 * DI;
    float* xc      = wf;             wf += (size_t)ROWS * DI;
    float* proj    = wf;             wf += (size_t)ROWS * 96;
    float* dt      = wf;             wf += (size_t)ROWS * DI;
    float* yb      = wf;             wf += (size_t)ROWS * DI;
    float* T       = wf;             wf += (size_t)ROWS * HID;
    float* logits  = wf;             wf += (size_t)ROWS * LATD;
    float* samples = wf;             wf += (size_t)ROWS * LATD;
    float* vfc     = wf;             wf += (size_t)ROWS * 4096;
    float* vd1     = wf;             wf += (size_t)ROWS * 32 * 16 * 16;
    float* vd2     = wf;             wf += (size_t)ROWS * 16 * 32 * 32;

    // 1. pack states
    pack_states_kernel<<<(ROWS * HID + 255) / 256, 256, 0, stream>>>(obs, actions, rewards, X);

    // 2. predictor stack: h = h + mamba(h)
    for (int i = 0; i < 8; ++i)
        run_mamba(pred[i], X, X, /*residual=*/X, xz, xc, proj, dt, yb, stream);

    // 3. emit h output
    copy_kernel<<<(ROWS * HID + 255) / 256, 256, 0, stream>>>(X, out_h, ROWS * HID);

    // 4. image head: t = mamba(h); logits = t @ img_W + img_b
    run_mamba(img_p, X, T, nullptr, xz, xc, proj, dt, yb, stream);
    launch_gemm_t<HID, LATD, HID, LATD, 0, true, false>(T, img_W, logits, img_b, nullptr, stream);

    // 5. softmax(24) + mix + ST-sample
    softmax_sample_kernel<<<(ROWS * CCAT + 255) / 256, 256, 0, stream>>>(logits, out_dists, samples);

    // 6. VAE decode
    launch_gemm_t<LATD, 4096, LATD, 4096, 1, true, false>(samples, fc_W, vfc, fc_b, nullptr, stream);
    {
        long t1 = (long)ROWS * 32 * 16 * 16;
        deconv_kernel<<<(int)((t1 + 255) / 256), 256, 0, stream>>>(vfc, dec_W0, dec_b0, vd1, 64, 32, 8, 8, 1);
        long t2 = (long)ROWS * 16 * 32 * 32;
        deconv_kernel<<<(int)((t2 + 255) / 256), 256, 0, stream>>>(vd1, dec_W1, dec_b1, vd2, 32, 16, 16, 16, 1);
        long t3 = (long)ROWS * 1 * 64 * 64;
        deconv_kernel<<<(int)((t3 + 255) / 256), 256, 0, stream>>>(vd2, dec_W2, dec_b2, out_obs, 16, 1, 32, 32, 3);
    }

    // 7. reward head: r = mamba(h); pred_rewards = r @ rew_W + rew_b
    run_mamba(rew_p, X, T, nullptr, xz, xc, proj, dt, yb, stream);
    rowdot_kernel<<<(ROWS * 32 + 255) / 256, 256, 0, stream>>>(T, rew_W, rew_b, out_rew, ROWS, HID);
}
